// SeparateRWGramAttention_43224550867973
// MI455X (gfx1250) — matrix operand order repaired and checked by reading, not compiled, not run on hardware
//
#include <hip/hip_runtime.h>
#include <hip/hip_bf16.h>
#include <stdint.h>

// ---------------- problem constants ----------------
constexpr int B_  = 2;
constexpr int T_  = 2048;
constexpr int D_  = 1024;
constexpr int H_  = 16;
constexpr int DH_ = 64;
constexpr int M_  = B_ * T_;   // 4096 GEMM rows
constexpr float SCALE_ = 0.125f; // DH^-0.5

typedef __bf16 bf16;
typedef __attribute__((ext_vector_type(16))) __bf16 v16bf;
typedef __attribute__((ext_vector_type(8)))  __bf16 v8bf;
typedef __attribute__((ext_vector_type(8)))  float  v8f;

#define WMMA_BF16(a, b, c) \
  __builtin_amdgcn_wmma_f32_16x16x32_bf16(false, (a), false, (b), (short)0, (c), false, false)

struct MaskTag   { static constexpr bool value = true;  };
struct NoMaskTag { static constexpr bool value = false; };

// Tensor Data Mover availability (probe-verified builtins)
#if __has_builtin(__builtin_amdgcn_tensor_load_to_lds) && __has_builtin(__builtin_amdgcn_s_wait_tensorcnt)
#define USE_TDM 1
#else
#define USE_TDM 0
#endif

#if USE_TDM
typedef unsigned int u32x4 __attribute__((ext_vector_type(4)));
typedef int          i32x4 __attribute__((ext_vector_type(4)));
typedef int          i32x8 __attribute__((ext_vector_type(8)));

__device__ __forceinline__ uint32_t lds_offset(const void* p) {
  return (uint32_t)(uintptr_t)(__attribute__((address_space(3))) const void*)p;
}

// Issue a TDM 2D tile load: `rows` x 32 cols of bf16, row stride K elems.
template<int K>
__device__ __forceinline__ void tdm_load_A(const bf16* gsrc, uint32_t lds_byte, int rows) {
  uint64_t gaddr = (uint64_t)(uintptr_t)gsrc;
  uint32_t glo = (uint32_t)gaddr, ghi = (uint32_t)(gaddr >> 32);
  u32x4 g0;
  g0[0] = (unsigned)__builtin_amdgcn_readfirstlane(1);              // count=1 (valid user D#)
  g0[1] = (unsigned)__builtin_amdgcn_readfirstlane((int)lds_byte);  // lds_addr (bytes)
  g0[2] = (unsigned)__builtin_amdgcn_readfirstlane((int)glo);       // global_addr[31:0]
  g0[3] = (unsigned)__builtin_amdgcn_readfirstlane(
              (int)((ghi & 0x01FFFFFFu) | (2u << 30)));             // addr[56:32] | type=2
  i32x8 g1;
  g1[0] = __builtin_amdgcn_readfirstlane(0x00010000);      // wg_mask=0, data_size=1 (2 bytes)
  g1[1] = __builtin_amdgcn_readfirstlane(32 << 16);        // tensor_dim0=32 (low16 at [31:16])
  g1[2] = __builtin_amdgcn_readfirstlane(rows << 16);      // dim0 hi=0 | tensor_dim1 lo=rows
  g1[3] = __builtin_amdgcn_readfirstlane(32 << 16);        // tensor_dim1 hi=0 | tile_dim0=32
  g1[4] = __builtin_amdgcn_readfirstlane(rows);            // tile_dim1=rows, tile_dim2=0
  g1[5] = __builtin_amdgcn_readfirstlane(K);               // tensor_dim0_stride low32 (elems)
  g1[6] = 0;
  g1[7] = 0;
  i32x4 z4 = {0, 0, 0, 0};
#if defined(__clang_major__) && __clang_major__ >= 23
  i32x8 z8 = {0, 0, 0, 0, 0, 0, 0, 0};
  __builtin_amdgcn_tensor_load_to_lds(g0, g1, z4, z4, z8, 0);
#else
  __builtin_amdgcn_tensor_load_to_lds(g0, g1, z4, z4, 0);
#endif
}
#endif // USE_TDM

// build a 16-element fragment from two contiguous 16B runs
__device__ __forceinline__ v16bf load_frag16(const bf16* p0, const bf16* p1) {
  v8bf lo = *(const v8bf*)p0;
  v8bf hi = *(const v8bf*)p1;
  v16bf r;
#pragma unroll
  for (int i = 0; i < 8; ++i) { r[i] = lo[i]; r[8 + i] = hi[i]; }
  return r;
}

// ---------------- conversion kernels ----------------
__global__ void f2bf_kernel(const float* __restrict__ src, bf16* __restrict__ dst, size_t n) {
  size_t i = (size_t)blockIdx.x * blockDim.x + threadIdx.x;
  if (i < n) dst[i] = (bf16)src[i];
}

__global__ void build_linew_kernel(const float* __restrict__ w1w, const float* __restrict__ w2w,
                                   const float* __restrict__ w1r, const float* __restrict__ w2r,
                                   bf16* __restrict__ dst) {
  int i = blockIdx.x * blockDim.x + threadIdx.x; // over D*64
  if (i >= D_ * 64) return;
  int r = i / 64, c = i % 64;
  dst[(size_t)r * 256 + c]       = (bf16)w1w[i];
  dst[(size_t)r * 256 + 64 + c]  = (bf16)w2w[i];
  dst[(size_t)r * 256 + 128 + c] = (bf16)w1r[i];
  dst[(size_t)r * 256 + 192 + c] = (bf16)w2r[i];
}

// ---------------- bf16 WMMA GEMM (double-buffered, pipelined) ---------------
// C(MxN) = A(MxK) @ B(KxN) (+ bias). Block = 8 waves, tile 256x64, BK=32.
// Each wave computes 32x64 (2 A-frags share every B-frag -> 8 WMMA / 12 loads).
// A tile: TDM direct-to-LDS (register-staged fallback). B tile: register-
// staged, stored transposed (n-major, stride 40) for contiguous B fragments.
constexpr int BKP = 40;

template<int N, int K>
__global__ __launch_bounds__(256) void gemm_bf16_kernel(
    const bf16* __restrict__ A, const bf16* __restrict__ Bm,
    const float* __restrict__ bias,
    float* __restrict__ Cf, bf16* __restrict__ Cb) {
  __shared__ bf16 As[2][256 * 32];
  __shared__ bf16 Bt[2][64 * BKP];
  const int tid  = threadIdx.x;
  const int wave = tid >> 5;
  const int lane = tid & 31;
  const int h    = lane >> 4;
  const int ln   = lane & 15;
  const int m0 = blockIdx.x * 256;
  const int n0 = blockIdx.y * 64;
  const int wm = wave * 32;     // wave row offset in tile
  const int bn  = tid & 63;     // B-staging: column
  const int bkg = tid >> 6;     // B-staging: k-group (0..3)

  v8f acc[2][4] = {};
  uint4 ra[4];
  bf16  rb[8];

  // ---- staging helpers ----
  auto issue_loads = [&](int k0) {
    const bf16* bp = &Bm[(size_t)(k0 + bkg * 8) * N + n0 + bn];
#pragma unroll
    for (int j = 0; j < 8; ++j) rb[j] = bp[(size_t)j * N];   // stride is compile-time imm
#if !USE_TDM
    const bf16* ap = &A[(size_t)(m0 + tid) * K + k0];        // thread = row
#pragma unroll
    for (int j = 0; j < 4; ++j) ra[j] = *(const uint4*)(ap + 8 * j);
#endif
  };
  auto commit_stage = [&](int buf) {
    *(uint4*)&Bt[buf][bn * BKP + bkg * 8] = *(const uint4*)rb;
#if !USE_TDM
    bf16* arow = &As[buf][tid * 32];
#pragma unroll
    for (int j = 0; j < 4; ++j) *(uint4*)(arow + 8 * j) = ra[j];
#endif
  };

  // ---- prologue: stage tile 0 ----
  issue_loads(0);
#if USE_TDM
  if (wave == 0) tdm_load_A<K>(&A[(size_t)m0 * K], lds_offset(&As[0][0]), 256);
#endif
  commit_stage(0);
#if USE_TDM
  if (wave == 0) __builtin_amdgcn_s_wait_tensorcnt(0);
#endif
  __syncthreads();

  int cur = 0;
  for (int k0 = 0; k0 < K; k0 += 32) {
    const bool nxt = (k0 + 32 < K);
    if (nxt) {
      issue_loads(k0 + 32);
#if USE_TDM
      if (wave == 0)
        tdm_load_A<K>(&A[(size_t)m0 * K + k0 + 32], lds_offset(&As[cur ^ 1][0]), 256);
#endif
      if (k0 + 64 < K) __builtin_prefetch(&A[(size_t)(m0 + tid) * K + k0 + 64], 0, 3);
    }

    // ---- compute from buf[cur]: 2 A-frags x 4 B-frags ----
    const int ar0 = (wm + ln) * 32;
    const int ar1 = (wm + 16 + ln) * 32;
    v16bf af0 = load_frag16(&As[cur][ar0 + h * 8], &As[cur][ar0 + 16 + h * 8]);
    v16bf af1 = load_frag16(&As[cur][ar1 + h * 8], &As[cur][ar1 + 16 + h * 8]);
#pragma unroll
    for (int nc = 0; nc < 4; ++nc) {
      const bf16* bp = &Bt[cur][(nc * 16 + ln) * BKP + h * 8];
      v16bf bfr = load_frag16(bp, bp + 16);
      acc[0][nc] = WMMA_BF16(af0, bfr, acc[0][nc]);
      acc[1][nc] = WMMA_BF16(af1, bfr, acc[1][nc]);
    }

    if (nxt) {
      commit_stage(cur ^ 1);
#if USE_TDM
      if (wave == 0) __builtin_amdgcn_s_wait_tensorcnt(0);
#endif
    }
    __syncthreads();
    cur ^= 1;
  }

  // ---- store: C layout — VGPR r holds row (r + 8h), col ln (+16*nc) ----
#pragma unroll
  for (int u = 0; u < 2; ++u)
#pragma unroll
    for (int nc = 0; nc < 4; ++nc) {
      int col = n0 + nc * 16 + ln;
      float bv = bias ? bias[col] : 0.f;
#pragma unroll
      for (int r = 0; r < 8; ++r) {
        int row = m0 + wm + u * 16 + r + 8 * h;
        float val = acc[u][nc][r] + bv;
        if (Cf) Cf[(size_t)row * N + col] = val;
        if (Cb) Cb[(size_t)row * N + col] = (bf16)val;
      }
    }
}

// ---------------- flash attention (1 wave per 16-row Q tile per (b,h)) ------
// V rows register-pipelined; K loads overlap the V scatter; one masked tile.
__global__ __launch_bounds__(32, 1) void attn_kernel(const bf16* __restrict__ qkv,
                                                     float* __restrict__ seqout) {
  __shared__ bf16  Ks[32 * 64];        // s-major (contiguous K fragments)
  __shared__ bf16  Vt[64 * BKP];       // transposed: Vt[n][k], contiguous V fragments
  __shared__ float Ps[16 * 32];
  const int lane = threadIdx.x;
  const int h = lane >> 4, ln = lane & 15;
  const int qt = blockIdx.x;
  const int b  = blockIdx.y / H_;
  const int hh = blockIdx.y % H_;
  const int t0 = qt * 16;
  const size_t rs = 3 * D_;
  const bf16* qbase = qkv + (size_t)b * T_ * rs + (size_t)hh * DH_;
  const bf16* kbase = qbase + D_;
  const bf16* vbase = qbase + 2 * D_;

  // Q fragments (two 16x32 K-slices), contiguous global runs per lane
  v16bf qf[2];
  {
    const bf16* qrow = &qbase[(size_t)(t0 + ln) * rs];
#pragma unroll
    for (int kk = 0; kk < 2; ++kk)
      qf[kk] = load_frag16(&qrow[kk * 32 + h * 8], &qrow[kk * 32 + 16 + h * 8]);
  }

  v8f O[4] = {};
  float mrow[8], lrow[8];
#pragma unroll
  for (int r = 0; r < 8; ++r) { mrow[r] = -3.0e38f; lrow[r] = 0.f; }

  // V row staging registers (one row per lane, pipelined across tiles)
  uint4 rv[8];
  auto fetch_v = [&](int sb) {
    const bf16* vrow = &vbase[(size_t)(sb + lane) * rs];
#pragma unroll
    for (int j = 0; j < 8; ++j) rv[j] = *(const uint4*)&vrow[8 * j];
  };
  auto commit_tile = [&](int sb) {
    const bf16* krow = &kbase[(size_t)(sb + lane) * rs];
    uint4 kk[8];
#pragma unroll
    for (int j = 0; j < 8; ++j) kk[j] = *(const uint4*)&krow[8 * j];  // K loads in flight
#pragma unroll
    for (int j = 0; j < 8; ++j) {                                     // V scatter overlaps
      const bf16* wb = (const bf16*)&rv[j];
#pragma unroll
      for (int i = 0; i < 8; ++i) Vt[(8 * j + i) * BKP + lane] = wb[i];
    }
#pragma unroll
    for (int j = 0; j < 8; ++j) *(uint4*)&Ks[lane * 64 + 8 * j] = kk[j];
  };

  auto compute_tile = [&](int sb, auto tag) {
    constexpr bool MASKED = decltype(tag)::value;
    // S = Q K^T, 16x32 in two 16x16 chunks
    float pv[2][8];
    float nmax[8];
#pragma unroll
    for (int r = 0; r < 8; ++r) nmax[r] = mrow[r];
#pragma unroll
    for (int c = 0; c < 2; ++c) {
      v8f sacc = {};
#pragma unroll
      for (int kk = 0; kk < 2; ++kk) {
        const bf16* kp = &Ks[(c * 16 + ln) * 64 + kk * 32 + h * 8];
        v16bf kf = load_frag16(kp, kp + 16);
        sacc = WMMA_BF16(qf[kk], kf, sacc);
      }
#pragma unroll
      for (int r = 0; r < 8; ++r) {
        float sv = sacc[r] * SCALE_;
        if (MASKED) {
          int s = sb + c * 16 + ln;
          int t = t0 + r + 8 * h;
          if (s > t) sv = -3.0e38f;
        }
        pv[c][r] = sv;
        nmax[r] = fmaxf(nmax[r], sv);
      }
    }
    // row-max across the 16 lanes that share a row (stays within lane-half)
#pragma unroll
    for (int r = 0; r < 8; ++r)
      for (int m = 1; m < 16; m <<= 1)
        nmax[r] = fmaxf(nmax[r], __shfl_xor(nmax[r], m, 32));

    float resc[8];
#pragma unroll
    for (int r = 0; r < 8; ++r) { resc[r] = __expf(mrow[r] - nmax[r]); mrow[r] = nmax[r]; }
#pragma unroll
    for (int nc = 0; nc < 4; ++nc)
#pragma unroll
      for (int r = 0; r < 8; ++r) O[nc][r] *= resc[r];

#pragma unroll
    for (int r = 0; r < 8; ++r) {
      float base = lrow[r] * resc[r];
      float psum = 0.f;
#pragma unroll
      for (int c = 0; c < 2; ++c) {
        float p = __expf(pv[c][r] - mrow[r]);
        psum += p;
        Ps[(r + 8 * h) * 32 + c * 16 + ln] = p;
      }
      for (int m = 1; m < 16; m <<= 1) psum += __shfl_xor(psum, m, 32);
      lrow[r] = base + psum;
    }
    __syncthreads();

    // P (16x32) @ V (32x64): P from LDS in A layout (contiguous f32 runs)
    v16bf pf;
#pragma unroll
    for (int g = 0; g < 2; ++g)
#pragma unroll
      for (int u = 0; u < 8; ++u)
        pf[g * 8 + u] = (bf16)Ps[ln * 32 + g * 16 + h * 8 + u];
#pragma unroll
    for (int nc = 0; nc < 4; ++nc) {
      const bf16* vp = &Vt[(nc * 16 + ln) * BKP + h * 8];
      v16bf vf = load_frag16(vp, vp + 16);
      O[nc] = WMMA_BF16(pf, vf, O[nc]);
    }
  };

  // ---- main loop: pipelined, mask-free tiles strictly below the diagonal ---
  const int nt = t0 / 32;        // number of mask-free 32-wide s-tiles
  if (nt > 0) fetch_v(0);
  for (int it = 0; it < nt; ++it) {
    __syncthreads();             // previous tile's LDS readers done
    commit_tile(32 * it);        // K direct (overlapped) + V from regs
    if (it + 1 < nt) {
      fetch_v(32 * (it + 1));    // overlap next V fetch with this tile's compute
      __builtin_prefetch(&kbase[(size_t)(32 * (it + 1) + lane) * rs], 0, 3);
      __builtin_prefetch(&vbase[(size_t)(32 * (it + 2) + lane) * rs], 0, 3);
    }
    __syncthreads();
    compute_tile(32 * it, NoMaskTag{});
  }

  // ---- single masked diagonal tile ----
  {
    int sb = 32 * nt;
    __syncthreads();
    int s = sb + lane;
    if (s < T_) {
      const bf16* krow = &kbase[(size_t)s * rs];
      const bf16* vrow = &vbase[(size_t)s * rs];
#pragma unroll
      for (int c0 = 0; c0 < 64; c0 += 8) {
        *(uint4*)&Ks[lane * 64 + c0] = *(const uint4*)&krow[c0];
        uint4 w = *(const uint4*)&vrow[c0];
        const bf16* wb = (const bf16*)&w;
#pragma unroll
        for (int i = 0; i < 8; ++i) Vt[(c0 + i) * BKP + lane] = wb[i];
      }
    } else {
      uint4 z = make_uint4(0u, 0u, 0u, 0u);
#pragma unroll
      for (int c0 = 0; c0 < 64; c0 += 8) {
        *(uint4*)&Ks[lane * 64 + c0] = z;
#pragma unroll
        for (int i = 0; i < 8; ++i) Vt[(c0 + i) * BKP + lane] = (bf16)0.f;
      }
    }
    __syncthreads();
    compute_tile(sb, MaskTag{});
  }

  // write seq_out (b, t, h*64+d) fp32
#pragma unroll
  for (int nc = 0; nc < 4; ++nc)
#pragma unroll
    for (int r = 0; r < 8; ++r) {
      int t = t0 + r + 8 * h;
      seqout[((size_t)b * T_ + t) * D_ + hh * DH_ + nc * 16 + ln] = O[nc][r] / lrow[r];
    }
}

// ---------------- Plucker lines + J6 transform ----------------
__device__ inline void exterior6(const float* p1, const float* p2, float* L) {
  L[0] = p1[0] * p2[1] - p1[1] * p2[0];
  L[1] = p1[0] * p2[2] - p1[2] * p2[0];
  L[2] = p1[0] * p2[3] - p1[3] * p2[0];
  L[3] = p1[1] * p2[2] - p1[2] * p2[1];
  L[4] = p1[1] * p2[3] - p1[3] * p2[1];
  L[5] = p1[2] * p2[3] - p1[3] * p2[2];
  float n = sqrtf(L[0]*L[0] + L[1]*L[1] + L[2]*L[2] + L[3]*L[3] + L[4]*L[4] + L[5]*L[5]);
  n = fmaxf(n, 1e-12f);
  for (int i = 0; i < 6; ++i) L[i] /= n;
}

__global__ void lines_kernel(const float* __restrict__ lp,
                             float* __restrict__ Jw, float* __restrict__ Jr,
                             float* __restrict__ rdl) {
  int idx = blockIdx.x * blockDim.x + threadIdx.x; // over B*T*H
  if (idx >= B_ * T_ * H_) return;
  int hh = idx % H_;
  int t  = (idx / H_) % T_;
  int b  = idx / (H_ * T_);
  const float* row  = lp + ((size_t)b * T_ + t) * 256;
  float p1[4], p2[4], L[6];
  size_t o = (((size_t)b * H_ + hh) * T_ + t) * 6;

#pragma unroll
  for (int i = 0; i < 4; ++i) {
    p1[i] = (t > 0) ? row[-256 + hh * 4 + i] : 0.f;
    p2[i] = row[64 + hh * 4 + i];
  }
  exterior6(p1, p2, L);
  Jw[o+0] = L[5]; Jw[o+1] = -L[4]; Jw[o+2] = L[3];
  Jw[o+3] = L[2]; Jw[o+4] = -L[1]; Jw[o+5] = L[0];

#pragma unroll
  for (int i = 0; i < 4; ++i) { p1[i] = row[128 + hh * 4 + i]; p2[i] = row[192 + hh * 4 + i]; }
  exterior6(p1, p2, L);
#pragma unroll
  for (int j = 0; j < 6; ++j) rdl[o + j] = L[j];
  Jr[o+0] = L[5]; Jr[o+1] = -L[4]; Jr[o+2] = L[3];
  Jr[o+3] = L[2]; Jr[o+4] = -L[1]; Jr[o+5] = L[0];
}

// ---------------- decayed Gram-state scan (one wave, barrier-free) ----------
__global__ __launch_bounds__(32) void scan_kernel(const float* __restrict__ Jw,
                                                  const float* __restrict__ Jr,
                                                  const float* __restrict__ rdl,
                                                  const float* __restrict__ decay_logits,
                                                  float* __restrict__ scores) {
  int g = blockIdx.x;
  int variant = g / (B_ * H_);
  int bh = g % (B_ * H_);
  int hh = bh % H_;
  float d = 1.f / (1.f + __expf(-decay_logits[hh]));
  const float* Jp = (variant ? Jr : Jw) + (size_t)bh * T_ * 6;
  const float* Rp = (variant ? Jw : rdl) + (size_t)bh * T_ * 6;
  float* out = scores + (size_t)variant * B_ * H_ * T_ + (size_t)bh * T_;

  const int lane = threadIdx.x;
  float Mrow[6] = {0.f, 0.f, 0.f, 0.f, 0.f, 0.f};
  for (int t = 0; t < T_; ++t) {
    float jv[6], rv[6];
#pragma unroll
    for (int u = 0; u < 6; ++u) { jv[u] = Jp[(size_t)t * 6 + u]; rv[u] = Rp[(size_t)t * 6 + u]; }
    float doti = 0.f;
#pragma unroll
    for (int u = 0; u < 6; ++u) doti += Mrow[u] * rv[u];
    float contrib = (lane < 6) ? rv[lane] * doti : 0.f;
    contrib += __shfl_xor(contrib, 1, 32);
    contrib += __shfl_xor(contrib, 2, 32);
    contrib += __shfl_xor(contrib, 4, 32);
    if (lane == 0) out[t] = contrib;
    float ji = (lane < 6) ? jv[lane] : 0.f;
#pragma unroll
    for (int u = 0; u < 6; ++u) Mrow[u] = d * (Mrow[u] + ji * jv[u]);
  }
}

// ---------------- gate + combine: comb = seq_out + gated * mem_val (bf16) ---
__global__ __launch_bounds__(128) void gate_combine_kernel(
    const float* __restrict__ x, const float* __restrict__ memgw,
    const float* __restrict__ memgb, const float* __restrict__ mem_scale,
    const float* __restrict__ rw_mix, const float* __restrict__ scores,
    const float* __restrict__ seqout, const float* __restrict__ memv,
    bf16* __restrict__ comb) {
  int bt = blockIdx.x;
  int b = bt / T_, t = bt % T_;
  int tid = threadIdx.x;
  int hh = tid >> 3, part = tid & 7;
  const float* xr = x + (size_t)bt * D_;

  float dot = 0.f;
  for (int dd = part * 128; dd < part * 128 + 128; ++dd)
    dot += xr[dd] * memgw[(size_t)dd * H_ + hh];
  dot += __shfl_xor(dot, 1, 32);
  dot += __shfl_xor(dot, 2, 32);
  dot += __shfl_xor(dot, 4, 32);

  __shared__ float sg[16];
  __shared__ float gmean_s;
  if (part == 0) sg[hh] = 1.f / (1.f + __expf(-(dot + memgb[hh])));
  __syncthreads();
  if (tid == 0) {
    float alpha = 1.f / (1.f + __expf(-rw_mix[0]));
    float acc = 0.f;
    for (int hx = 0; hx < H_; ++hx) {
      size_t so = ((size_t)b * H_ + hx) * T_ + t;
      float ms = (1.f - alpha) * scores[so] + alpha * scores[(size_t)B_ * H_ * T_ + so];
      acc += (1.f / (1.f + __expf(-ms * mem_scale[hx]))) * sg[hx];
    }
    gmean_s = acc / (float)H_;
  }
  __syncthreads();
  float g = gmean_s;
  for (int dc = tid; dc < D_; dc += 128) {
    size_t o = (size_t)bt * D_ + dc;
    comb[o] = (bf16)(seqout[o] + g * memv[o]);
  }
}

// ---------------- host launcher ----------------
extern "C" void kernel_launch(void* const* d_in, const int* in_sizes, int n_in,
                              void* d_out, int out_size, void* d_ws, size_t ws_size,
                              hipStream_t stream) {
  const float* x       = (const float*)d_in[0];
  const float* qkv_w   = (const float*)d_in[1];
  const float* qkv_b   = (const float*)d_in[2];
  const float* w1w     = (const float*)d_in[3];
  const float* w2w     = (const float*)d_in[4];
  const float* w1r     = (const float*)d_in[5];
  const float* w2r     = (const float*)d_in[6];
  const float* memv_w  = (const float*)d_in[7];
  const float* memv_b  = (const float*)d_in[8];
  const float* memg_w  = (const float*)d_in[9];
  const float* memg_b  = (const float*)d_in[10];
  const float* memsc   = (const float*)d_in[11];
  const float* rw_mix  = (const float*)d_in[12];
  const float* out_w   = (const float*)d_in[13];
  const float* out_b   = (const float*)d_in[14];
  const float* decay_l = (const float*)d_in[15];
  float* outp = (float*)d_out;

  char* ws = (char*)d_ws;
  size_t off = 0;
  auto alloc = [&](size_t bytes) {
    void* p = ws + off;
    off += (bytes + 255) & ~(size_t)255;
    return p;
  };
  bf16*  x_bf     = (bf16*) alloc((size_t)M_ * D_ * 2);
  bf16*  qkvw_bf  = (bf16*) alloc((size_t)D_ * 3 * D_ * 2);
  bf16*  qkv_bf   = (bf16*) alloc((size_t)M_ * 3 * D_ * 2);
  bf16*  linew_bf = (bf16*) alloc((size_t)D_ * 256 * 2);
  float* lineproj = (float*)alloc((size_t)M_ * 256 * 4);
  bf16*  memvw_bf = (bf16*) alloc((size_t)D_ * D_ * 2);
  float* memv_f   = (float*)alloc((size_t)M_ * D_ * 4);
  bf16*  outw_bf  = (bf16*) alloc((size_t)D_ * D_ * 2);
  float* seqout   = (float*)alloc((size_t)M_ * D_ * 4);
  float* Jw       = (float*)alloc((size_t)B_ * H_ * T_ * 6 * 4);
  float* Jr       = (float*)alloc((size_t)B_ * H_ * T_ * 6 * 4);
  float* rdl      = (float*)alloc((size_t)B_ * H_ * T_ * 6 * 4);
  float* scores   = (float*)alloc((size_t)2 * B_ * H_ * T_ * 4);
  bf16*  comb_bf  = (bf16*) alloc((size_t)M_ * D_ * 2);
  (void)ws_size; (void)n_in; (void)in_sizes; (void)out_size;

  auto cvt = [&](const float* s, bf16* dst, size_t n) {
    f2bf_kernel<<<dim3((unsigned)((n + 255) / 256)), dim3(256), 0, stream>>>(s, dst, n);
  };
  cvt(x, x_bf, (size_t)M_ * D_);
  cvt(qkv_w, qkvw_bf, (size_t)D_ * 3 * D_);
  cvt(memv_w, memvw_bf, (size_t)D_ * D_);
  cvt(out_w, outw_bf, (size_t)D_ * D_);
  build_linew_kernel<<<dim3(D_ * 64 / 256), dim3(256), 0, stream>>>(w1w, w2w, w1r, w2r, linew_bf);

  // QKV GEMM (bf16 out for attention)
  gemm_bf16_kernel<3 * D_, D_><<<dim3(M_ / 256, 3 * D_ / 64), dim3(256), 0, stream>>>(
      x_bf, qkvw_bf, qkv_b, nullptr, qkv_bf);
  // line projections (fp32 out)
  gemm_bf16_kernel<256, D_><<<dim3(M_ / 256, 256 / 64), dim3(256), 0, stream>>>(
      x_bf, linew_bf, nullptr, lineproj, nullptr);
  // mem value projection (fp32 out)
  gemm_bf16_kernel<D_, D_><<<dim3(M_ / 256, D_ / 64), dim3(256), 0, stream>>>(
      x_bf, memvw_bf, memv_b, memv_f, nullptr);

  // attention
  attn_kernel<<<dim3(T_ / 16, B_ * H_), dim3(32), 0, stream>>>(qkv_bf, seqout);

  // lines + scans + gate
  lines_kernel<<<dim3(B_ * T_ * H_ / 128), dim3(128), 0, stream>>>(lineproj, Jw, Jr, rdl);
  scan_kernel<<<dim3(2 * B_ * H_), dim3(32), 0, stream>>>(Jw, Jr, rdl, decay_l, scores);
  gate_combine_kernel<<<dim3(B_ * T_), dim3(128), 0, stream>>>(
      x, memg_w, memg_b, memsc, rw_mix, scores, seqout, memv_f, comb_bf);

  // final output GEMM -> d_out (fp32)
  gemm_bf16_kernel<D_, D_><<<dim3(M_ / 256, D_ / 64), dim3(256), 0, stream>>>(
      comb_bf, outw_bf, out_b, outp, nullptr);
}
